// GraphAttentionLayer_1279900254938
// MI455X (gfx1250) — matrix operand order repaired and checked by reading, not compiled, not run on hardware
//
#include <hip/hip_runtime.h>
#include <hip/hip_bf16.h>

typedef _Float16 half_t;
typedef __attribute__((ext_vector_type(16))) _Float16 v16h;
typedef __attribute__((ext_vector_type(8)))  _Float16 v8h;
typedef __attribute__((ext_vector_type(8)))  float    v8f;

#define BS 4
#define NN 512
#define DD 128
#define ROWS (BS * NN)            // 2048
#define XE (BS * NN * DD)         // 262144
#define TRI_PAIRS 528             // #(k,jt) tile pairs with jt>=k, 32 tiles
#define WAVES_PER_B (TRI_PAIRS * 16)   // 8448
#define TOTAL_WAVES (BS * WAVES_PER_B) // 33792

// Hardware tanh: single V_TANH_F32 TRANS op (co-issues with WMMA) instead of
// the branchy libm expansion.
__device__ __forceinline__ float fast_tanh(float x) {
#if __has_builtin(__builtin_amdgcn_tanhf)
    return __builtin_amdgcn_tanhf(x);
#else
    float r;
    asm volatile("v_tanh_f32 %0, %1" : "=v"(r) : "v"(x));
    return r;
#endif
}

// ---------------------------------------------------------------------------
// Kernel 1: x (fp32) -> xh (f16)
// ---------------------------------------------------------------------------
__global__ __launch_bounds__(256) void k_convert_x(const float* __restrict__ x,
                                                   half_t* __restrict__ xh) {
    int idx = blockIdx.x * 256 + threadIdx.x;
    if (idx < XE) xh[idx] = (half_t)x[idx];
}

// ---------------------------------------------------------------------------
// Kernel 2: att_proj_w (d,o) fp32 -> WhT (o,d) f16  (transposed for B-frag)
// ---------------------------------------------------------------------------
__global__ __launch_bounds__(256) void k_convert_wt(const float* __restrict__ W,
                                                    half_t* __restrict__ whT) {
    int idx = blockIdx.x * 256 + threadIdx.x;    // 16384
    int d = idx >> 7, o = idx & 127;             // coalesced read over o
    whT[o * 128 + d] = (half_t)W[idx];
}

// ---------------------------------------------------------------------------
// Kernel 3: scores[b,i,j] = sum_o tanh( sum_d x_id x_jd W_do + bias_o ) * aw_o
// Symmetric in (i,j): enumerate only upper-triangular j-tiles (jt >= i/16)
// and mirror-store. One wave per (b, i, j-tile): GEMM M=16(j) N=128(o) K=128(d)
// -> 8 o-tiles x 4 k-steps = 32 x v_wmma_f32_16x16x32_f16 per wave.
// ---------------------------------------------------------------------------
__global__ __launch_bounds__(256) void k_scores(const half_t* __restrict__ xh,
                                                const half_t* __restrict__ whTg,
                                                const float* __restrict__ bias,
                                                const float* __restrict__ aw,
                                                float* __restrict__ scores) {
    __shared__ half_t WhT[128 * 128];            // 32 KB: W^T [o][d] in f16
    const int tid = threadIdx.x;

    // cooperative 32KB stage (b128 copies)
    {
        const uint4* src = (const uint4*)whTg;
        uint4* dst = (uint4*)WhT;
        #pragma unroll
        for (int k = 0; k < 8; ++k) dst[tid + 256 * k] = src[tid + 256 * k];
    }
    __syncthreads();

    const int lane = tid & 31;
    const int wave = tid >> 5;
    const int g    = blockIdx.x * 8 + wave;      // 33792 waves total
    const int b    = g / WAVES_PER_B;
    const int w    = g - b * WAVES_PER_B;
    const int p    = w >> 4;                     // triangular pair 0..527
    const int il   = w & 15;

    // decode p -> (k, jt) with jt >= k;   offset(k) = 32k - k(k-1)/2
    int k = (int)((65.0f - sqrtf(4225.0f - 8.0f * (float)p)) * 0.5f);
    while (32 * (k + 1) - ((k + 1) * k) / 2 <= p) ++k;     // float-rounding fixups
    while (32 * k - (k * (k - 1)) / 2 > p) --k;
    const int jt    = k + (p - (32 * k - (k * (k - 1)) / 2));
    const int i     = k * 16 + il;
    const int jbase = jt * 16;

    const int mrow = lane & 15;                  // A-matrix row (j_local) / B col (o_local)
    const int hi   = lane >> 4;                  // lane half
    const half_t* xi = xh + (size_t)(b * NN + i) * DD;
    const half_t* xj = xh + (size_t)(b * NN + jbase + mrow) * DD;

    // Build all 4 A fragments: A[m=j][k=d] = x_j[d] * x_i[d]
    // Contiguous k order (not ISA table order; B uses the same): lane half 'hi' holds contiguous d-run [d0+16*hi .. +16)
    v16h A[4];
    #pragma unroll
    for (int kk = 0; kk < 4; ++kk) {
        const int d0 = kk * 32 + hi * 16;
        v8h aj0 = *(const v8h*)(xj + d0);
        v8h aj1 = *(const v8h*)(xj + d0 + 8);
        v8h ai0 = *(const v8h*)(xi + d0);
        v8h ai1 = *(const v8h*)(xi + d0 + 8);
        #pragma unroll
        for (int e = 0; e < 8; ++e) {
            A[kk][e]     = aj0[e] * ai0[e];
            A[kk][8 + e] = aj1[e] * ai1[e];
        }
    }

    float part[8] = {0.f, 0.f, 0.f, 0.f, 0.f, 0.f, 0.f, 0.f};
    const int ocol = mrow;
    #pragma unroll
    for (int t = 0; t < 8; ++t) {                // o-tiles
        v8f acc = {};
        // B[k=d][n=o]: lane's 16 halves are contiguous d at W^T[o][kk*32 + 16*hi ..]
        const half_t* wrow = WhT + (t * 16 + ocol) * 128 + hi * 16;
        #pragma unroll
        for (int kk = 0; kk < 4; ++kk) {
            v16h B = *(const v16h*)(wrow + kk * 32);
            acc = __builtin_amdgcn_wmma_f32_16x16x32_f16(
                false, A[kk], false, B, (short)0, acc, false, false);
        }
        const int   o  = t * 16 + ocol;
        const float bo = bias[o];
        const float wo = aw[o];
        #pragma unroll
        for (int r = 0; r < 8; ++r)
            part[r] = fmaf(fast_tanh(acc[r] + bo), wo, part[r]);
    }

    // reduce over the 16 o-lanes (stays within each lane-half)
    #pragma unroll
    for (int m = 1; m < 16; m <<= 1) {
        #pragma unroll
        for (int r = 0; r < 8; ++r) part[r] += __shfl_xor(part[r], m, 32);
    }
    if (mrow == 0) {                             // lanes 0 (rows 0-7) and 16 (rows 8-15)
        const int rbase = hi * 8;
        float* srow = scores + ((size_t)(b * NN + i) * NN) + jbase + rbase;
        #pragma unroll
        for (int r = 0; r < 8; ++r) srow[r] = part[r];
        // symmetric mirror: scores[b][j][i] = scores[b][i][j] (bit-identical
        // under i<->j, so diagonal-tile double-writes are benign)
        float* scol = scores + ((size_t)(b * NN + jbase + rbase) * NN) + i;
        #pragma unroll
        for (int r = 0; r < 8; ++r) scol[(size_t)r * NN] = part[r];
    }
}

// ---------------------------------------------------------------------------
// Kernel 4: per (b,i) row: softmax over j, then agg[b,i,d] = sum_j att_j x[b,j,d]
// ---------------------------------------------------------------------------
__global__ __launch_bounds__(256) void k_softmax_agg(const float* __restrict__ scores,
                                                     const float* __restrict__ x,
                                                     float* __restrict__ agg) {
    __shared__ float satt[NN];
    __shared__ float sred[256];
    const int row = blockIdx.x;                  // b*512 + i
    const int b   = row >> 9;
    const int tid = threadIdx.x;
    const float* s = scores + (size_t)row * NN;

    float v0 = s[tid], v1 = s[tid + 256];
    sred[tid] = fmaxf(v0, v1);
    __syncthreads();
    for (int w = 128; w > 0; w >>= 1) {
        if (tid < w) sred[tid] = fmaxf(sred[tid], sred[tid + w]);
        __syncthreads();
    }
    const float m = sred[0];
    __syncthreads();
    float e0 = __expf(v0 - m), e1 = __expf(v1 - m);
    sred[tid] = e0 + e1;
    __syncthreads();
    for (int w = 128; w > 0; w >>= 1) {
        if (tid < w) sred[tid] += sred[tid + w];
        __syncthreads();
    }
    const float inv = 1.0f / sred[0];
    satt[tid] = e0 * inv;
    satt[tid + 256] = e1 * inv;
    __syncthreads();

    const int d = tid & 127;
    const int h = tid >> 7;                      // split j-space in two
    const float* xb = x + (size_t)b * NN * DD;
    float acc = 0.f;
    for (int j = h; j < NN; j += 2) acc = fmaf(satt[j], xb[j * DD + d], acc);
    sred[tid] = acc;
    __syncthreads();
    if (tid < 128) agg[(size_t)row * DD + tid] = sred[tid] + sred[tid + 128];
}

// ---------------------------------------------------------------------------
// Kernel 5: y = agg @ W1 + b1 + x @ W2 + b2   (one block per row)
// ---------------------------------------------------------------------------
__global__ __launch_bounds__(128) void k_proj(const float* __restrict__ agg,
                                              const float* __restrict__ x,
                                              const float* __restrict__ W1,
                                              const float* __restrict__ b1,
                                              const float* __restrict__ W2,
                                              const float* __restrict__ b2,
                                              float* __restrict__ y) {
    __shared__ float ra[DD], rx[DD];
    const int row = blockIdx.x;
    const int o   = threadIdx.x;
    ra[o] = agg[(size_t)row * DD + o];
    rx[o] = x[(size_t)row * DD + o];
    __syncthreads();
    float acc = b1[o] + b2[o];
    #pragma unroll 8
    for (int d = 0; d < DD; ++d)
        acc = fmaf(ra[d], W1[d * DD + o], fmaf(rx[d], W2[d * DD + o], acc));
    y[(size_t)row * DD + o] = acc;
}

// ---------------------------------------------------------------------------
// Kernel 6: BatchNorm stats over 2048 rows (biased var), one block of 128
// ---------------------------------------------------------------------------
__global__ __launch_bounds__(128) void k_bn_stats(const float* __restrict__ y,
                                                  float* __restrict__ mean,
                                                  float* __restrict__ rstd) {
    const int o = threadIdx.x;
    float s = 0.f, s2 = 0.f;
    for (int r = 0; r < ROWS; ++r) {
        float v = y[(size_t)r * DD + o];
        s += v;
        s2 = fmaf(v, v, s2);
    }
    const float mu  = s * (1.0f / ROWS);
    const float var = s2 * (1.0f / ROWS) - mu * mu;
    mean[o] = mu;
    rstd[o] = rsqrtf(var + 1e-5f);
}

// ---------------------------------------------------------------------------
// Kernel 7: BN apply + SELU
// ---------------------------------------------------------------------------
__global__ __launch_bounds__(256) void k_bn_selu(const float* __restrict__ y,
                                                 const float* __restrict__ mean,
                                                 const float* __restrict__ rstd,
                                                 const float* __restrict__ gamma,
                                                 const float* __restrict__ beta,
                                                 float* __restrict__ out) {
    const int idx = blockIdx.x * 256 + threadIdx.x;   // 262144
    const int o   = idx & 127;
    float v = (y[idx] - mean[o]) * rstd[o] * gamma[o] + beta[o];
    const float kScale = 1.0507009873554805f;
    const float kAlpha = 1.6732632423543772f;
    out[idx] = v > 0.f ? kScale * v : kScale * kAlpha * (__expf(v) - 1.0f);
}

// ---------------------------------------------------------------------------
extern "C" void kernel_launch(void* const* d_in, const int* in_sizes, int n_in,
                              void* d_out, int out_size, void* d_ws, size_t ws_size,
                              hipStream_t stream) {
    const float* x     = (const float*)d_in[0];  // (4,512,128)
    const float* apw   = (const float*)d_in[1];  // att_proj_w (128,128)
    const float* apb   = (const float*)d_in[2];  // att_proj_b (128)
    const float* aw    = (const float*)d_in[3];  // att_weight (128)
    const float* pw_w  = (const float*)d_in[4];  // proj_with_w
    const float* pw_b  = (const float*)d_in[5];
    const float* po_w  = (const float*)d_in[6];  // proj_without_w
    const float* po_b  = (const float*)d_in[7];
    const float* gamma = (const float*)d_in[8];
    const float* beta  = (const float*)d_in[9];
    float* out = (float*)d_out;

    // workspace layout (256B aligned regions)
    char* ws = (char*)d_ws;
    half_t* xh     = (half_t*)(ws + 0);                      // 512 KB
    half_t* whT    = (half_t*)(ws + 524288);                 // 32 KB
    float*  scores = (float*)(ws + 557056);                  // 4 MB
    float*  agg    = (float*)(ws + 4751360);                 // 1 MB
    float*  y      = (float*)(ws + 5799936);                 // 1 MB
    float*  mean   = (float*)(ws + 6848512);                 // 512 B
    float*  rstd   = (float*)(ws + 6849024);                 // 512 B

    k_convert_x<<<XE / 256, 256, 0, stream>>>(x, xh);
    k_convert_wt<<<64, 256, 0, stream>>>(apw, whT);
    // 33792 upper-triangular waves, 8 waves per block
    k_scores<<<TOTAL_WAVES / 8, 256, 0, stream>>>(xh, whT, apb, aw, scores);
    k_softmax_agg<<<ROWS, 256, 0, stream>>>(scores, x, agg);
    k_proj<<<ROWS, 128, 0, stream>>>(agg, x, pw_w, pw_b, po_w, po_b, y);
    k_bn_stats<<<1, 128, 0, stream>>>(y, mean, rstd);
    k_bn_selu<<<XE / 256, 256, 0, stream>>>(y, mean, rstd, gamma, beta, out);
}
